// idm_11948599017799
// MI455X (gfx1250) — matrix operand order repaired and checked by reading, not compiled, not run on hardware
//
#include <hip/hip_runtime.h>
#include <hip/hip_bf16.h>

// ---------------------------------------------------------------------------
// LIIF-style implicit upsampler on MI455X (gfx1250, wave32, WMMA + TDM).
// Compute-bound: ~2.1e11 FLOPs of MLP GEMMs vs ~70MB of HBM traffic -> run the
// MLP on v_wmma_f32_16x16x32_f16 (f16 in, f32 accumulate). All packed weights
// (208 KB) are staged into the WGP's 320 KB LDS once per block via the Tensor
// Data Mover, so the steady state is pure LDS->WMMA with no L2 weight refetch.
// ---------------------------------------------------------------------------

typedef _Float16 h16;
typedef __attribute__((ext_vector_type(16))) _Float16 v16h;
typedef __attribute__((ext_vector_type(8)))  _Float16 v8h;
typedef __attribute__((ext_vector_type(8)))  float    v8f;
typedef __attribute__((ext_vector_type(4)))  unsigned int v4u;
typedef __attribute__((ext_vector_type(8)))  int v8i;
typedef __attribute__((ext_vector_type(4)))  int v4i;
typedef unsigned int u32;

#define BATCH   4
#define CCH     64
#define HH      64
#define WWID    64
#define QOUT    65536        // 256*256 output queries
#define K1P     96           // 66 (=C+2) padded up to 3*32
#define HID     256
#define TILE_M  64           // queries per workgroup

// Packed-weight (fragment-major) layout, all halves.
// Fragment = one 32(K) x 16(N) B-tile: 32 lanes x 16 halves = 512 halves.
// Lane mapping (contiguous k order, same as the A loader): ncol = lane&15, g = lane>>4,
// half h (0..15) holds K = g*16 + h  ->  one contiguous 32B run per lane.
#define FRAG_H     512
#define W1P_FRAGS  (3*16)    // K1P/32 kc  x  HID/16 nt
#define W2P_FRAGS  (8*16)    // 256/32     x  256/16
#define W3P_FRAGS  (8*4)     // 256/32     x  64/16
#define W1P_OFF    0
#define W2P_OFF    (W1P_FRAGS*FRAG_H)
#define W3P_OFF    (W2P_OFF + W2P_FRAGS*FRAG_H)
#define WPACK_TOTAL (W3P_OFF + W3P_FRAGS*FRAG_H)   // 106496 halves = 212992 B
#define WPACK_BYTES (WPACK_TOTAL*2)

// Dynamic LDS: [ weight pack | sIn tile | sH tile ]
#define SMEM_BYTES (WPACK_BYTES + TILE_M*K1P*2 + TILE_M*HID*2)   // 258048 B

#if __has_builtin(__builtin_amdgcn_tensor_load_to_lds)
#define HAVE_TDM 1
#endif

// ---------------------------------------------------------------------------
// Kernel 0: f32 -> f16 weight conversion + fragment-major repack (runs once).
// ---------------------------------------------------------------------------
__global__ void repack_weights(const float* __restrict__ W1,
                               const float* __restrict__ W2,
                               const float* __restrict__ W3,
                               h16* __restrict__ wp) {
    int t = blockIdx.x * blockDim.x + threadIdx.x;
    if (t >= WPACK_TOTAL) return;
    int e    = t & (FRAG_H - 1);
    int lane = e >> 4;
    int hh   = e & 15;
    int g    = lane >> 4;
    int ncol = lane & 15;
    float v = 0.0f;
    if (t < W2P_OFF) {                       // W1: [66 x 256], K padded to 96
        int frag = t >> 9;
        int kc = frag >> 4, nt = frag & 15;
        int K = kc * 32 + g * 16 + hh;
        if (K < CCH + 2) v = W1[K * HID + nt * 16 + ncol];
    } else if (t < W3P_OFF) {                // W2: [256 x 256]
        int frag = (t - W2P_OFF) >> 9;
        int kc = frag >> 4, nt = frag & 15;
        int K = kc * 32 + g * 16 + hh;
        v = W2[K * HID + nt * 16 + ncol];
    } else {                                 // W3: [256 x 64]
        int frag = (t - W3P_OFF) >> 9;
        int kc = frag >> 2, nt = frag & 3;
        int K = kc * 32 + g * 16 + hh;
        v = W3[K * CCH + nt * 16 + ncol];
    }
    wp[t] = (h16)v;
}

// ---------------------------------------------------------------------------
// Fragment loaders
// ---------------------------------------------------------------------------
// A-fragment (16x32, 16-bit) from row-major LDS. Lane (m = lane&15, g=lane>>4)
// needs K = [16g, 16g+16): two contiguous v8h (b128) loads.
__device__ __forceinline__ v16h load_a_frag(const h16* base, int ldk, int lane, int kc) {
    int m = lane & 15, g = lane >> 4;
    const h16* p = base + (size_t)m * ldk + kc * 32 + g * 16;
    v8h lo = *(const v8h*)p;
    v8h hi = *(const v8h*)(p + 8);
    v16h a;
#pragma unroll
    for (int i = 0; i < 8; ++i) { a[i] = lo[i]; a[i + 8] = hi[i]; }
    return a;
}

// B-fragment: contiguous 32B per lane from the fragment-major pack (now LDS).
__device__ __forceinline__ v16h load_b_frag(const h16* pack, int ntiles, int kc,
                                            int nt, int lane) {
    const h16* p = pack + (size_t)(kc * ntiles + nt) * FRAG_H + lane * 16;
    v8h lo = *(const v8h*)p;
    v8h hi = *(const v8h*)(p + 8);
    v16h b;
#pragma unroll
    for (int i = 0; i < 8; ++i) { b[i] = lo[i]; b[i + 8] = hi[i]; }
    return b;
}

__device__ __forceinline__ v8f vzero8() {
    v8f z;
#pragma unroll
    for (int i = 0; i < 8; ++i) z[i] = 0.0f;
    return z;
}

#define WMMA_F16(A, B, C) \
    __builtin_amdgcn_wmma_f32_16x16x32_f16(false, (A), false, (B), (short)0, (C), false, false)

// ---------------------------------------------------------------------------
// Kernel 1: TDM weight staging + fused gather + 3-layer WMMA MLP + blend.
// Block = 256 threads (8 waves). Each block: 64 queries x 1 batch x 4 corners.
// ---------------------------------------------------------------------------
__global__ void __launch_bounds__(256)
liif_mlp_wmma(const float* __restrict__ x,
              const float* __restrict__ b1,
              const float* __restrict__ b2,
              const float* __restrict__ b3,
              const h16*  __restrict__ wp,
              float* __restrict__ out) {
    extern __shared__ __align__(16) char smemRaw[];
    h16* sW  = (h16*)smemRaw;                               // 208 KB weight pack
    h16* sIn = (h16*)(smemRaw + WPACK_BYTES);               // 12 KB layer-1 A tile
    h16* sH  = sIn + TILE_M * K1P;                          // 32 KB hidden tile

    __shared__ int   sSidx[4 * 64];
    __shared__ float sRelY[4 * 64], sRelX[4 * 64], sArea[4 * 64], sInvTot[64];

    const int tid  = threadIdx.x;
    const int b    = blockIdx.x >> 10;        // 1024 tiles per batch element
    const int tile = blockIdx.x & 1023;
    const int lane = tid & 31;
    const int wv   = tid >> 5;
    const int ncol = lane & 15;
    const int g    = lane >> 4;
    const float* xb = x + (size_t)b * CCH * HH * WWID;

    // ---- Stage the packed weights into LDS with the Tensor Data Mover ----
#ifdef HAVE_TDM
    if (wv == 0) {
        const u32 lds_off = (u32)(size_t)sW;        // addr[31:0] = LDS offset
        const unsigned long long ga = (unsigned long long)(size_t)wp;
        const u32 UNITS = WPACK_BYTES / 8;          // 26624 8-byte elements
        v4u g0;
        g0[0] = 1u;                                  // count=1, user-mode, no gather
        g0[1] = lds_off;                             // lds_addr
        g0[2] = (u32)ga;                             // global_addr[31:0]
        g0[3] = (u32)((ga >> 32) & 0x1FFFFFFull) | (2u << 30); // addr[56:32], type=2
        v8i g1;
        g1[0] = (int)(3u << 16);                     // wg_mask=0, data_size=3 (8B)
        g1[1] = (int)(UNITS << 16);                  // tensor_dim0[15:0], atomic_addr=0
        g1[2] = (int)(1u << 16);                     // tensor_dim1=1, tensor_dim0[31:16]=0
        g1[3] = (int)(UNITS << 16);                  // tile_dim0=UNITS
        g1[4] = 1;                                   // tile_dim1=1, tile_dim2=0
        g1[5] = (int)UNITS;                          // tensor_dim0_stride[31:0]
        g1[6] = 0;
        g1[7] = 0;
        v4i z4; z4[0] = 0; z4[1] = 0; z4[2] = 0; z4[3] = 0;
#if defined(__clang_major__) && __clang_major__ >= 23
        v8i z8; z8[0]=0; z8[1]=0; z8[2]=0; z8[3]=0; z8[4]=0; z8[5]=0; z8[6]=0; z8[7]=0;
        __builtin_amdgcn_tensor_load_to_lds(g0, g1, z4, z4, z8, 0);
#else
        __builtin_amdgcn_tensor_load_to_lds(g0, g1, z4, z4, 0);
#endif
    }
#else
    // Fallback: cooperative vector copy global -> LDS.
    for (int i = tid; i < WPACK_TOTAL / 8; i += 256)
        ((v8h*)sW)[i] = ((const v8h*)wp)[i];
#endif
    if (tid == 0) __builtin_prefetch(xb, 0, 0);

    // ---- Geometry for all 4 corners (thread = (corner, row)), overlaps TDM ----
    {
        int ci = tid >> 6, r = tid & 63;
        int q  = tile * 64 + r;
        int qy = q >> 8, qx = q & 255;
        float cy = -1.0f + 1.0f / 256.0f + (2.0f / 256.0f) * (float)qy;
        float cx = -1.0f + 1.0f / 256.0f + (2.0f / 256.0f) * (float)qx;
        float vx = (ci < 2) ? -1.0f : 1.0f;   // outer loop in reference
        float vy = (ci & 1) ? 1.0f : -1.0f;   // inner loop in reference
        const float rr = 1.0f / 64.0f, lim = 1.0f - 1e-6f;
        float sy = fminf(fmaxf(cy + vx * rr + 1e-6f, -lim), lim);
        float sx = fminf(fmaxf(cx + vy * rr + 1e-6f, -lim), lim);
        float fiy = fminf(fmaxf(rintf(((sy + 1.0f) * 64.0f - 1.0f) * 0.5f), 0.0f), 63.0f);
        float fix = fminf(fmaxf(rintf(((sx + 1.0f) * 64.0f - 1.0f) * 0.5f), 0.0f), 63.0f);
        int iy = (int)fiy, ix = (int)fix;
        float fy = -1.0f + 1.0f / 64.0f + (2.0f / 64.0f) * (float)iy;
        float fx = -1.0f + 1.0f / 64.0f + (2.0f / 64.0f) * (float)ix;
        float ry_ = (cy - fy) * 64.0f;
        float rx_ = (cx - fx) * 64.0f;
        sSidx[ci * 64 + r] = iy * WWID + ix;
        sRelY[ci * 64 + r] = ry_;
        sRelX[ci * 64 + r] = rx_;
        sArea[ci * 64 + r] = fabsf(ry_ * rx_) + 1e-9f;
    }
#ifdef HAVE_TDM
    if (wv == 0) __builtin_amdgcn_s_wait_tensorcnt(0);  // weights landed in LDS
#endif
    __syncthreads();
    if (tid < 64)
        sInvTot[tid] = 1.0f / (sArea[tid] + sArea[64 + tid] + sArea[128 + tid] + sArea[192 + tid]);
    __syncthreads();

    // Persistent weighted-output accumulators: wave wv owns M-tile (wv>>1),
    // N-tiles {(wv&1)*2, (wv&1)*2+1} of the 64x64 output tile.
    v8f ret[2];
    ret[0] = vzero8();
    ret[1] = vzero8();
    const int mtw = wv >> 1;

    for (int ci = 0; ci < 4; ++ci) {
        // ---- Gather features + rel coords into padded f16 input tile ----
        for (int idx = tid; idx < TILE_M * K1P; idx += 256) {
            int r = idx / K1P;
            int c = idx - r * K1P;
            float v;
            if (c < CCH)      v = xb[c * (HH * WWID) + sSidx[ci * 64 + r]];
            else if (c == 64) v = sRelY[ci * 64 + r];
            else if (c == 65) v = sRelX[ci * 64 + r];
            else              v = 0.0f;
            sIn[idx] = (h16)v;
        }
        __syncthreads();

        // ---- Layer 1: [64 x 96] x [96 x 256] -> relu -> sH ----
#pragma unroll
        for (int ntl = 0; ntl < 2; ++ntl) {
            int nt = wv * 2 + ntl;
            float bias = b1[nt * 16 + ncol];
#pragma unroll
            for (int mt = 0; mt < 4; ++mt) {
                v8f acc = vzero8();
#pragma unroll
                for (int kc = 0; kc < 3; ++kc) {
                    v16h a  = load_a_frag(sIn + mt * 16 * K1P, K1P, lane, kc);
                    v16h bf = load_b_frag(sW + W1P_OFF, 16, kc, nt, lane);
                    acc = WMMA_F16(a, bf, acc);
                }
#pragma unroll
                for (int i = 0; i < 8; ++i) {
                    int row = mt * 16 + i + 8 * g;      // CDNA5 C/D layout
                    sH[row * HID + nt * 16 + ncol] = (h16)fmaxf(acc[i] + bias, 0.0f);
                }
            }
        }
        __syncthreads();

        // ---- Layer 2 compute: [64 x 256] x [256 x 256] (results in regs) ----
        v8f acc2[8];
#pragma unroll
        for (int ntl = 0; ntl < 2; ++ntl) {
            int nt = wv * 2 + ntl;
#pragma unroll
            for (int mt = 0; mt < 4; ++mt) {
                v8f acc = vzero8();
#pragma unroll
                for (int kc = 0; kc < 8; ++kc) {
                    v16h a  = load_a_frag(sH + mt * 16 * HID, HID, lane, kc);
                    v16h bf = load_b_frag(sW + W2P_OFF, 16, kc, nt, lane);
                    acc = WMMA_F16(a, bf, acc);
                }
                acc2[ntl * 4 + mt] = acc;
            }
        }
        __syncthreads();        // all reads of sH done before overwrite

        // ---- Layer 2 bias+relu -> sH ----
#pragma unroll
        for (int ntl = 0; ntl < 2; ++ntl) {
            int nt = wv * 2 + ntl;
            float bias = b2[nt * 16 + ncol];
#pragma unroll
            for (int mt = 0; mt < 4; ++mt) {
#pragma unroll
                for (int i = 0; i < 8; ++i) {
                    int row = mt * 16 + i + 8 * g;
                    sH[row * HID + nt * 16 + ncol] =
                        (h16)fmaxf(acc2[ntl * 4 + mt][i] + bias, 0.0f);
                }
            }
        }
        __syncthreads();

        // ---- Layer 3: [64 x 256] x [256 x 64], area-weighted accumulate ----
#pragma unroll
        for (int j = 0; j < 2; ++j) {
            int nt = (wv & 1) * 2 + j;
            float bias = b3[nt * 16 + ncol];
            v8f acc = vzero8();
#pragma unroll
            for (int kc = 0; kc < 8; ++kc) {
                v16h a  = load_a_frag(sH + mtw * 16 * HID, HID, lane, kc);
                v16h bf = load_b_frag(sW + W3P_OFF, 4, kc, nt, lane);
                acc = WMMA_F16(a, bf, acc);
            }
#pragma unroll
            for (int i = 0; i < 8; ++i) {
                int row = mtw * 16 + i + 8 * g;
                // local-ensemble swap: pred_ci weighted by area of corner 3-ci
                float wc = sArea[(3 - ci) * 64 + row] * sInvTot[row];
                ret[j][i] += (acc[i] + bias) * wc;
            }
        }
        // next corner's post-gather barrier separates these sH reads from the
        // next layer-1 writes, so no extra barrier here.
    }

    // ---- Write blended output [B, Q, 64] ----
#pragma unroll
    for (int j = 0; j < 2; ++j) {
        int nt = (wv & 1) * 2 + j;
#pragma unroll
        for (int i = 0; i < 8; ++i) {
            int row = mtw * 16 + i + 8 * g;
            size_t q = (size_t)tile * 64 + row;
            out[((size_t)b * QOUT + q) * CCH + nt * 16 + ncol] = ret[j][i];
        }
    }
}

// ---------------------------------------------------------------------------
extern "C" void kernel_launch(void* const* d_in, const int* in_sizes, int n_in,
                              void* d_out, int out_size, void* d_ws, size_t ws_size,
                              hipStream_t stream) {
    const float* x  = (const float*)d_in[0];
    const float* W1 = (const float*)d_in[1];
    const float* b1 = (const float*)d_in[2];
    const float* W2 = (const float*)d_in[3];
    const float* b2 = (const float*)d_in[4];
    const float* W3 = (const float*)d_in[5];
    const float* b3 = (const float*)d_in[6];
    (void)in_sizes; (void)n_in; (void)out_size; (void)ws_size;

    h16* wp = (h16*)d_ws;   // needs WPACK_BYTES = 212,992 bytes of scratch

    // Allow >64KB dynamic LDS (gfx1250 WGP has 320KB). Harmless if a no-op.
    (void)hipFuncSetAttribute((const void*)liif_mlp_wmma,
                              hipFuncAttributeMaxDynamicSharedMemorySize,
                              SMEM_BYTES);

    repack_weights<<<(WPACK_TOTAL + 255) / 256, 256, 0, stream>>>(W1, W2, W3, wp);
    liif_mlp_wmma<<<BATCH * (QOUT / TILE_M), 256, SMEM_BYTES, stream>>>(
        x, b1, b2, b3, wp, (float*)d_out);
}
